// SelfAttentionBlock_13408887898356
// MI455X (gfx1250) — matrix operand order repaired and checked by reading, not compiled, not run on hardware
//
#include <hip/hip_runtime.h>
#include <hip/hip_bf16.h>
#include <stdint.h>

// ---------------------------------------------------------------------------
// Types for CDNA5 WMMA (gfx1250, wave32)
// ---------------------------------------------------------------------------
typedef __attribute__((ext_vector_type(16))) __bf16 v16bf;
typedef __attribute__((ext_vector_type(8)))  float  v8f;

union PackAB { v16bf v; unsigned int u[8]; };

#define D_MODEL  1024
#define N_HEADS  16
#define HEAD_DIM 64

// f32 -> bf16 (round to nearest even), pure bit manipulation
__device__ __forceinline__ unsigned short f2bf(float f) {
  unsigned int u = __float_as_uint(f);
  unsigned int r = u + 0x7FFFu + ((u >> 16) & 1u);
  return (unsigned short)(r >> 16);
}

// ISA 16-bit A-matrix 16x32 layout: vgpr r, lane-half -> K index (even, pairs)
__device__ __forceinline__ int kOfA(int r, int half) {
  return ((r >> 2) << 4) + (half << 3) + ((r & 3) << 1);
}
// ISA 16-bit B-matrix 32x16 layout: vgpr r, lane-half -> K (row) index
__device__ __forceinline__ int kOfB(int r, int half) {
  return ((r >> 2) << 4) + (half << 3) + ((r & 3) << 1);
}

// ---------------------------------------------------------------------------
// 1) Pre-LayerNorm: one wave per token, f32 in -> bf16 out
// ---------------------------------------------------------------------------
__global__ __launch_bounds__(32, 1)
void ln_kernel(const float* __restrict__ x,
               const float* __restrict__ g,
               const float* __restrict__ b,
               unsigned short* __restrict__ h,
               int n_tok) {
  int row = blockIdx.x;
  if (row >= n_tok) return;
  int lane = threadIdx.x;
  const float* xr = x + (size_t)row * D_MODEL;
  float vals[32];
  float s = 0.f, s2 = 0.f;
  for (int i = 0; i < 32; ++i) {
    float v = xr[lane + 32 * i];
    vals[i] = v; s += v; s2 += v * v;
  }
  for (int m = 16; m; m >>= 1) {
    s  += __shfl_xor(s,  m, 32);
    s2 += __shfl_xor(s2, m, 32);
  }
  float mu  = s  * (1.0f / 1024.0f);
  float var = s2 * (1.0f / 1024.0f) - mu * mu;
  float rs  = rsqrtf(var + 1e-5f);
  unsigned short* hr = h + (size_t)row * D_MODEL;
  for (int i = 0; i < 32; ++i) {
    int c = lane + 32 * i;
    hr[c] = f2bf((vals[i] - mu) * rs * g[c] + b[c]);
  }
}

// ---------------------------------------------------------------------------
// 2) f32 -> bf16 weight conversion
// ---------------------------------------------------------------------------
__global__ void cvt_kernel(const float* __restrict__ in,
                           unsigned short* __restrict__ out, int n) {
  int i = blockIdx.x * blockDim.x + threadIdx.x;
  if (i < n) out[i] = f2bf(in[i]);
}

// ---------------------------------------------------------------------------
// 3) WMMA GEMM: C[M,N] = A[M,1024] * W[N,1024]^T (+bias [+x residual])
//    MODE 0: scatter to q (tok-major), k (tok-major), v (dim-major/transposed)
//    MODE 1: fuse out_b + residual x, write f32 d_out
//    Wave tile: 32(M) x 64(N); block = 4 waves -> 32 x 256
// ---------------------------------------------------------------------------
template <int MODE>
__global__ __launch_bounds__(128, 1)
void gemm_wmma(const unsigned short* __restrict__ A,
               const unsigned short* __restrict__ W,
               const float* __restrict__ bias,
               const float* __restrict__ xres,
               unsigned short* __restrict__ qb,
               unsigned short* __restrict__ kb,
               unsigned short* __restrict__ vT,
               float* __restrict__ outf,
               int Mtok) {
  int lane = threadIdx.x & 31;
  int wave = threadIdx.x >> 5;
  int half = lane >> 4;
  int n16  = lane & 15;
  int mbase = blockIdx.x * 32;
  int nbase = blockIdx.y * 256 + wave * 64;

  v8f c[2][4];
  for (int rt = 0; rt < 2; ++rt)
    for (int ct = 0; ct < 4; ++ct)
      for (int p = 0; p < 8; ++p) c[rt][ct][p] = 0.0f;

  int rowA0 = min(mbase + n16,       Mtok - 1);
  int rowA1 = min(mbase + 16 + n16,  Mtok - 1);
  const unsigned short* a0p = A + (size_t)rowA0 * D_MODEL;
  const unsigned short* a1p = A + (size_t)rowA1 * D_MODEL;
  const unsigned short* wp0 = W + (size_t)(nbase + n16) * D_MODEL;

  for (int k0 = 0; k0 < D_MODEL; k0 += 32) {
    PackAB a0, a1;
#pragma unroll
    for (int r = 0; r < 8; ++r) {
      int kd = k0 + kOfA(r, half);
      a0.u[r] = *(const unsigned int*)(a0p + kd);
      a1.u[r] = *(const unsigned int*)(a1p + kd);
    }
    if (k0 + 32 < D_MODEL) {
      __builtin_prefetch(a0p + k0 + 32, 0, 1);
      __builtin_prefetch(wp0 + k0 + 32, 0, 1);
    }
#pragma unroll
    for (int ct = 0; ct < 4; ++ct) {
      const unsigned short* wp = wp0 + (size_t)(ct * 16) * D_MODEL;
      PackAB bb;
#pragma unroll
      for (int r = 0; r < 8; ++r) {
        int kd = k0 + kOfB(r, half);
        bb.u[r] = *(const unsigned int*)(wp + kd);
      }
      c[0][ct] = __builtin_amdgcn_wmma_f32_16x16x32_bf16(
          false, a0.v, false, bb.v, (short)0, c[0][ct], false, false);
      c[1][ct] = __builtin_amdgcn_wmma_f32_16x16x32_bf16(
          false, a1.v, false, bb.v, (short)0, c[1][ct], false, false);
    }
  }

  // Epilogue. C layout: lane holds col = n16, rows p + 8*half.
  for (int ct = 0; ct < 4; ++ct) {
    int col = nbase + ct * 16 + n16;
    float bv = bias[col];
    for (int rt = 0; rt < 2; ++rt) {
      int row0 = mbase + rt * 16 + half * 8;
      if (MODE == 0 && col >= 2048 && row0 + 7 < Mtok) {
        // v region, full row-tile: rows are contiguous in vT -> packed store
        unsigned int pk[4];
#pragma unroll
        for (int j = 0; j < 4; ++j) {
          unsigned int lo = f2bf(c[rt][ct][2 * j]     + bv);
          unsigned int hi = f2bf(c[rt][ct][2 * j + 1] + bv);
          pk[j] = lo | (hi << 16);
        }
        unsigned short* dst = vT + (size_t)(col - 2048) * Mtok + row0;
        ((uint2*)dst)[0] = make_uint2(pk[0], pk[1]);
        ((uint2*)dst)[1] = make_uint2(pk[2], pk[3]);
        continue;
      }
#pragma unroll
      for (int p = 0; p < 8; ++p) {
        int row = row0 + p;
        if (row >= Mtok) continue;
        float val = c[rt][ct][p] + bv;
        if (MODE == 0) {
          if (col < 1024)
            qb[(size_t)row * D_MODEL + col] = f2bf(val);
          else if (col < 2048)
            kb[(size_t)row * D_MODEL + (col - 1024)] = f2bf(val);
          else
            vT[(size_t)(col - 2048) * Mtok + row] = f2bf(val);
        } else {
          val += xres[(size_t)row * D_MODEL + col];
          outf[(size_t)row * D_MODEL + col] = val;
        }
      }
    }
  }
}

// ---------------------------------------------------------------------------
// 4) Flash-style attention: one wave per (16-query tile, head), ragged batches
//    K tile staged to LDS via CDNA5 async copy (ASYNCcnt path)
// ---------------------------------------------------------------------------
__global__ __launch_bounds__(32, 1)
void attn_kernel(const unsigned short* __restrict__ qb,
                 const unsigned short* __restrict__ kbuf,
                 const unsigned short* __restrict__ vT,
                 const int* __restrict__ offsets,
                 unsigned short* __restrict__ ob,
                 int Mtok, int nBatch) {
  __shared__ float lp[16][33];              // P tile staging (C -> A layout)
  __shared__ unsigned short ktile[32 * 64]; // 32 keys x 64 dims, bf16 (4 KB)
  int lane = threadIdx.x;
  int half = lane >> 4;
  int n16  = lane & 15;
  int head = blockIdx.y;

  // map linear q-tile -> (batch, tile)
  int rem = blockIdx.x;
  int batch = -1, base = 0, L = 0, tile = 0;
  for (int b = 0; b < nBatch; ++b) {
    int s = offsets[b], e = offsets[b + 1];
    int len = e - s;
    int nt = (len + 15) >> 4;
    if (rem < nt) { batch = b; tile = rem; base = s; L = len; break; }
    rem -= nt;
  }
  if (batch < 0) return;
  int q0 = tile * 16;

  // Q tile as two A operands (head-dim 0-31, 32-63)
  PackAB aq[2];
  int qrowc = min(q0 + n16, L - 1);
  const unsigned short* qp =
      qb + (size_t)(base + qrowc) * D_MODEL + head * HEAD_DIM;
#pragma unroll
  for (int cc = 0; cc < 2; ++cc)
#pragma unroll
    for (int r = 0; r < 8; ++r)
      aq[cc].u[r] = *(const unsigned int*)(qp + cc * 32 + kOfA(r, half));

  float mrow[8], lsum[8], alpha[8];
  v8f oacc[4];
  for (int p = 0; p < 8; ++p) { mrow[p] = -1e30f; lsum[p] = 0.0f; }
  for (int c4 = 0; c4 < 4; ++c4)
    for (int p = 0; p < 8; ++p) oacc[c4][p] = 0.0f;

  int nkb = (L + 31) >> 5;
  for (int kbI = 0; kbI < nkb; ++kbI) {
    int j0 = kbI * 32;

    // ---- async-stage K tile (32 keys x 64 dims bf16) into LDS ----
    // 8 issues x 32 lanes x 16B = 4 KB; tracked on ASYNCcnt.
#pragma unroll
    for (int i = 0; i < 8; ++i) {
      int elem = (i * 32 + lane) * 8;       // bf16 elements, 16B chunks
      int key  = elem >> 6;
      int kd   = elem & 63;
      int keyc = min(j0 + key, L - 1);
      const unsigned short* src =
          kbuf + (size_t)(base + keyc) * D_MODEL + head * HEAD_DIM + kd;
      unsigned int ldsoff =
          (unsigned int)(uintptr_t)(&ktile[(size_t)key * 64 + kd]);
      asm volatile("global_load_async_to_lds_b128 %0, %1, off"
                   :: "v"(ldsoff),
                      "v"((unsigned long long)(uintptr_t)src)
                   : "memory");
    }
    asm volatile("s_wait_asynccnt 0x0" ::: "memory");

    // ---- scores: S[t] = Q (16x64) @ K_t^T (64x16), scaled + masked ----
    v8f S[2];
    for (int t = 0; t < 2; ++t) {
      int key = j0 + t * 16 + n16;
      const unsigned short* kp = &ktile[(size_t)(t * 16 + n16) * 64];
      v8f s;
      for (int p = 0; p < 8; ++p) s[p] = 0.0f;
#pragma unroll
      for (int cc = 0; cc < 2; ++cc) {
        PackAB bb;
#pragma unroll
        for (int r = 0; r < 8; ++r)
          bb.u[r] = *(const unsigned int*)(kp + cc * 32 + kOfB(r, half));
        s = __builtin_amdgcn_wmma_f32_16x16x32_bf16(
            false, aq[cc].v, false, bb.v, (short)0, s, false, false);
      }
      bool masked = key >= L;
#pragma unroll
      for (int p = 0; p < 8; ++p)
        S[t][p] = masked ? -1e30f : s[p] * 0.125f;   // 1/sqrt(64)
    }

    // ---- online softmax (row m lives on vgpr p=m&7, lane-half m>>3) ----
#pragma unroll
    for (int p = 0; p < 8; ++p) {
      float mx = fmaxf(S[0][p], S[1][p]);
      for (int m = 8; m; m >>= 1) mx = fmaxf(mx, __shfl_xor(mx, m, 32));
      float mnew = fmaxf(mrow[p], mx);
      float a  = __expf(mrow[p] - mnew);
      float p0 = __expf(S[0][p] - mnew);
      float p1 = __expf(S[1][p] - mnew);
      float rs = p0 + p1;
      for (int m = 8; m; m >>= 1) rs += __shfl_xor(rs, m, 32);
      lsum[p] = lsum[p] * a + rs;
      mrow[p] = mnew;
      alpha[p] = a;
      lp[p + half * 8][n16]      = p0;
      lp[p + half * 8][16 + n16] = p1;
    }
    __syncthreads();

    // rescale O accumulators
    for (int c4 = 0; c4 < 4; ++c4)
#pragma unroll
      for (int p = 0; p < 8; ++p) oacc[c4][p] *= alpha[p];

    // P (16x32) re-laid-out as WMMA A operand via LDS
    PackAB ap;
#pragma unroll
    for (int r = 0; r < 8; ++r) {
      int kd = kOfA(r, half);
      unsigned int lo = f2bf(lp[n16][kd]);
      unsigned int hi = f2bf(lp[n16][kd + 1]);
      ap.u[r] = lo | (hi << 16);
    }
    __syncthreads();

    // ---- O += P (16x32) @ V (32x64); V is dim-major so pairs are dwords ----
    for (int c4 = 0; c4 < 4; ++c4) {
      const unsigned short* vp =
          vT + (size_t)(head * HEAD_DIM + c4 * 16 + n16) * Mtok;
      PackAB bv;
#pragma unroll
      for (int r = 0; r < 8; ++r) {
        int tok = min(base + j0 + kOfB(r, half), Mtok - 2);
        bv.u[r] = *(const unsigned int*)(vp + tok);
      }
      oacc[c4] = __builtin_amdgcn_wmma_f32_16x16x32_bf16(
          false, ap.v, false, bv.v, (short)0, oacc[c4], false, false);
    }
  }

  // finalize: divide by row sums, write valid rows (bf16, token-major)
#pragma unroll
  for (int p = 0; p < 8; ++p) {
    int row = q0 + p + half * 8;
    if (row >= L) continue;
    float inv = lsum[p] > 0.0f ? 1.0f / lsum[p] : 0.0f;
    for (int c4 = 0; c4 < 4; ++c4)
      ob[(size_t)(base + row) * D_MODEL + head * HEAD_DIM + c4 * 16 + n16] =
          f2bf(oacc[c4][p] * inv);
  }
}

// ---------------------------------------------------------------------------
// Launch
// ---------------------------------------------------------------------------
extern "C" void kernel_launch(void* const* d_in, const int* in_sizes, int n_in,
                              void* d_out, int out_size, void* d_ws,
                              size_t ws_size, hipStream_t stream) {
  const float* x     = (const float*)d_in[0];
  const int*   offs  = (const int*)d_in[1];
  const float* w_in  = (const float*)d_in[2];
  const float* b_in  = (const float*)d_in[3];
  const float* w_out = (const float*)d_in[4];
  const float* b_out = (const float*)d_in[5];
  const float* ln_g  = (const float*)d_in[6];
  const float* ln_b  = (const float*)d_in[7];

  int Mtok = in_sizes[0] / D_MODEL;
  int nB   = in_sizes[1] - 1;

  char* ws = (char*)d_ws;
  size_t tokb = (size_t)Mtok * D_MODEL * sizeof(unsigned short);
  unsigned short* h   = (unsigned short*)ws; ws += (tokb + 255) & ~(size_t)255;
  unsigned short* wq  = (unsigned short*)ws; ws += 3u * 1024u * 1024u * 2u;
  unsigned short* wo  = (unsigned short*)ws; ws += 1024u * 1024u * 2u;
  unsigned short* qb  = (unsigned short*)ws; ws += (tokb + 255) & ~(size_t)255;
  unsigned short* kb  = (unsigned short*)ws; ws += (tokb + 255) & ~(size_t)255;
  unsigned short* vT  = (unsigned short*)ws; ws += (tokb + 255) & ~(size_t)255;
  unsigned short* ob  = (unsigned short*)ws; ws += (tokb + 255) & ~(size_t)255;

  cvt_kernel<<<(3 * 1024 * 1024 + 255) / 256, 256, 0, stream>>>(
      w_in, wq, 3 * 1024 * 1024);
  cvt_kernel<<<(1024 * 1024 + 255) / 256, 256, 0, stream>>>(
      w_out, wo, 1024 * 1024);

  ln_kernel<<<Mtok, 32, 0, stream>>>(x, ln_g, ln_b, h, Mtok);

  dim3 g1((Mtok + 31) / 32, (3 * D_MODEL) / 256);
  gemm_wmma<0><<<g1, 128, 0, stream>>>(h, wq, b_in, nullptr, qb, kb, vT,
                                       nullptr, Mtok);

  dim3 g2((Mtok + 15) / 16 + nB, N_HEADS);
  attn_kernel<<<g2, 32, 0, stream>>>(qb, kb, vT, offs, ob, Mtok, nB);

  dim3 g3((Mtok + 31) / 32, D_MODEL / 256);
  gemm_wmma<1><<<g3, 128, 0, stream>>>(ob, wo, b_out, x, nullptr, nullptr,
                                       nullptr, (float*)d_out, Mtok);
}
